// mLSTMCell_31533649888019
// MI455X (gfx1250) — matrix operand order repaired and checked by reading, not compiled, not run on hardware
//
#include <hip/hip_runtime.h>
#include <hip/hip_bf16.h>

// ---------------------------------------------------------------------------
// mLSTM (B=8, L=2048, D=1024, DH=128) — chunkwise-parallel formulation so the
// sequential recurrence becomes bf16 WMMA GEMMs. All big FLOPs go through
// v_wmma_f32_16x16x32_bf16 with fp32 accumulation. The two large GEMMs
// (o-projection epilogue and q/k/v projection) are fed by the Tensor Data
// Mover (tensor_load_to_lds, TENSORcnt) with double-buffered LDS tiles.
// Transposed operands are staged through LDS so all fragments load as b128.
// ---------------------------------------------------------------------------

typedef __attribute__((ext_vector_type(16))) __bf16 v16bf;
typedef __attribute__((ext_vector_type(8)))  float  v8f;
typedef __attribute__((ext_vector_type(4)))  unsigned int v4u;
typedef __attribute__((ext_vector_type(4)))  int    v4i;
typedef __attribute__((ext_vector_type(8)))  int    v8i;

#if __has_builtin(__builtin_amdgcn_tensor_load_to_lds) && \
    __has_builtin(__builtin_amdgcn_s_wait_tensorcnt)
#define USE_TDM 1
#else
#define USE_TDM 0
#endif

__device__ inline v8f zero8() {
  v8f z = {0.f, 0.f, 0.f, 0.f, 0.f, 0.f, 0.f, 0.f};
  return z;
}

__device__ inline v8f wmma_bf16(v16bf a, v16bf b, v8f c) {
  // 8 args: (neg_a, A, neg_b, B, c_mod, C, reuse_a, reuse_b)
  return __builtin_amdgcn_wmma_f32_16x16x32_bf16(false, a, false, b, (short)0, c,
                                                 false, false);
}

union BF16x16 {
  uint4 q[2];
  v16bf v;
};

// A fragment (16x32 bf16): lane l holds row m=l&15; K elements
//   e in [0,8)  -> K = koff + e          (koff = (l>>4)*8)
//   e in [8,16) -> K = 16 + koff + (e-8)
__device__ inline v16bf load_a_fast(const __bf16* base, size_t stride) {
  int l = threadIdx.x & 31;
  const __bf16* p = base + (size_t)(l & 15) * stride + (size_t)((l >> 4) * 8);
  BF16x16 u;
  u.q[0] = *(const uint4*)(p);
  u.q[1] = *(const uint4*)(p + 16);
  return u.v;
}

// B fragment (32x16 bf16): lane l holds column n=l&15; K same as A fragment
//   (e<8 -> (l>>4)*8 + e; e>=8 -> 16 + (l>>4)*8 + (e-8)).
__device__ inline v16bf load_b_fast(const __bf16* base, size_t stride) {
  int l = threadIdx.x & 31;
  const __bf16* p = base + (size_t)(l & 15) * stride + (size_t)((l >> 4) * 8);
  BF16x16 u;
  u.q[0] = *(const uint4*)(p);
  u.q[1] = *(const uint4*)(p + 16);
  return u.v;
}

// ---------------------------------------------------------------------------
// Tensor Data Mover: 2-D tile (tile_k elems x tile_rows) from a row-major
// bf16 matrix (row stride = stride_elems) into LDS, packed row-major with
// LDS row stride = tile_k. Descriptor per CDNA5 ISA 8.3-8.6.
// ---------------------------------------------------------------------------
__device__ inline unsigned lds_offset_of(const void* p) {
  // Flat LDS addresses carry the LDS byte offset in the low 32 bits.
  return (unsigned)(unsigned long long)p;
}

__device__ inline void tdm_load_2d(const __bf16* gsrc, unsigned lds_off,
                                   unsigned tile_k, unsigned tile_rows,
                                   unsigned stride_elems) {
#if USE_TDM
  unsigned long long ga = (unsigned long long)gsrc;
  v4u g0;
  g0[0] = 1u;                                   // count=1, user descriptor
  g0[1] = lds_off;                              // lds_addr (bytes)
  g0[2] = (unsigned)(ga & 0xFFFFFFFFu);         // global_addr[31:0]
  g0[3] = (unsigned)((ga >> 32) & 0x01FFFFFFu)  // global_addr[56:32]
          | (2u << 30);                         // type=2 ("image")
  unsigned td0 = stride_elems;                  // tensor_dim0 (row length)
  unsigned td1 = tile_rows;                     // tensor_dim1
  v8i g1;
  g1[0] = 0x00010000;                           // data_size=1 -> 2 bytes
  g1[1] = (int)((td0 & 0xFFFFu) << 16);         // tensor_dim0[15:0]
  g1[2] = (int)(((td0 >> 16) & 0xFFFFu) | ((td1 & 0xFFFFu) << 16));
  g1[3] = (int)(((td1 >> 16) & 0xFFFFu) | ((tile_k & 0xFFFFu) << 16));
  g1[4] = (int)(tile_rows & 0xFFFFu);           // tile_dim1 (tile_dim2=0)
  g1[5] = (int)stride_elems;                    // tensor_dim0_stride[31:0]
  g1[6] = 0;                                    // stride[47:32], dim1_stride lo
  g1[7] = 0;
  v4i g2 = {0, 0, 0, 0};
  v4i g3 = {0, 0, 0, 0};
#if defined(__clang_major__) && (__clang_major__ >= 23)
  v8i g4 = {0, 0, 0, 0, 0, 0, 0, 0};
  __builtin_amdgcn_tensor_load_to_lds(g0, g1, g2, g3, g4, 0);
#else
  __builtin_amdgcn_tensor_load_to_lds(g0, g1, g2, g3, 0);
#endif
#else
  (void)gsrc; (void)lds_off; (void)tile_k; (void)tile_rows; (void)stride_elems;
#endif
}

// Stage a tile into LDS: TDM path = wave 0 issues DMA; fallback = cooperative
// vector copy by all 256 threads.
__device__ inline void stage_tile(const __bf16* gsrc, __bf16* lds,
                                  unsigned tile_k, unsigned tile_rows,
                                  unsigned stride_elems, int tid) {
#if USE_TDM
  if (tid < 32) tdm_load_2d(gsrc, lds_offset_of(lds), tile_k, tile_rows,
                            stride_elems);
#else
  int total = (int)(tile_rows * tile_k) / 8;    // uint4 count
  int kq = (int)tile_k / 8;
  for (int i = tid; i < total; i += 256) {
    int r = i / kq, c = i % kq;
    ((uint4*)lds)[i] = ((const uint4*)(gsrc + (size_t)r * stride_elems))[c];
  }
#endif
}

__device__ inline void stage_commit(int tid) {
#if USE_TDM
  if (tid < 32) __builtin_amdgcn_s_wait_tensorcnt(0);
#endif
  __syncthreads();
}

// Stage a 128x128 row-major bf16 chunk into LDS *transposed* with padded row
// stride 136 (coalesced global reads; fragments then load as ds b128).
__device__ inline void stage_transpose_128(const __bf16* __restrict__ src,
                                           __bf16* __restrict__ dst, int tid) {
  for (int i = tid; i < 128 * 128; i += 256) {
    int s = i >> 7, d = i & 127;          // consecutive tids -> consecutive d
    dst[d * 136 + s] = src[(size_t)s * 128 + d];
  }
}

// ---------------------------------------------------------------------------
// Kernels
// ---------------------------------------------------------------------------

__global__ void cvt_bf16_kernel(const float* __restrict__ src,
                                __bf16* __restrict__ dst, int n) {
  int i = blockIdx.x * blockDim.x + threadIdx.x;
  if (i < n) dst[i] = (__bf16)src[i];
}

// Y = X(16384x1024) @ Wqkv^T(384x1024) + bias -> q/k/v bf16 (N x 128 each).
// 128x128 output block per 256-thread block, TDM-staged, double buffered.
// Column-block 0/1/2 maps exactly onto the q/k/v weight segments.
__global__ void __launch_bounds__(256)
proj_qkv_kernel(const __bf16* __restrict__ xb, const __bf16* __restrict__ wqkv,
                const float* __restrict__ bq, const float* __restrict__ bk,
                const float* __restrict__ bv, __bf16* __restrict__ qb,
                __bf16* __restrict__ kb, __bf16* __restrict__ vb) {
  const int D = 1024, DH = 128, KT = 64;
  __shared__ __align__(16) __bf16 Abuf[2][128 * KT];  // 2 x 16 KB
  __shared__ __align__(16) __bf16 Bbuf[2][128 * KT];  // 2 x 16 KB
  int tid = threadIdx.x, wave = tid >> 5, l = tid & 31;
  int bm = blockIdx.x / 3, cb = blockIdx.x % 3;  // 128 row-blocks x 3 col-blocks
  int row0 = bm * 128, col0 = cb * 128;

  stage_tile(xb + (size_t)row0 * D, Abuf[0], KT, 128, D, tid);
  stage_tile(wqkv + (size_t)col0 * D, Bbuf[0], KT, 128, D, tid);
  stage_commit(tid);

  v8f acc[8];
#pragma unroll
  for (int i = 0; i < 8; ++i) acc[i] = zero8();

  const int KSTEPS = D / KT;  // 16
  for (int ks = 0; ks < KSTEPS; ++ks) {
    int cur = ks & 1;
    if (ks + 1 < KSTEPS) {
      int kk = (ks + 1) * KT;
      stage_tile(xb + (size_t)row0 * D + kk, Abuf[cur ^ 1], KT, 128, D, tid);
      stage_tile(wqkv + (size_t)col0 * D + kk, Bbuf[cur ^ 1], KT, 128, D, tid);
    }
    const __bf16* Ab = Abuf[cur] + wave * 16 * KT;
#pragma unroll
    for (int k2 = 0; k2 < KT; k2 += 32) {
      v16bf a = load_a_fast(Ab + k2, KT);
#pragma unroll
      for (int nt = 0; nt < 8; ++nt) {
        v16bf b = load_b_fast(Bbuf[cur] + nt * 16 * KT + k2, KT);
        acc[nt] = wmma_bf16(a, b, acc[nt]);
      }
    }
    stage_commit(tid);
  }

  // Block-uniform output/bias select (col-block == q/k/v segment).
  const float* bias = (cb == 0) ? bq : (cb == 1) ? bk : bv;
  __bf16* outp = (cb == 0) ? qb : (cb == 1) ? kb : vb;
  int mh = (l >> 4) * 8, n = l & 15;
#pragma unroll
  for (int nt = 0; nt < 8; ++nt) {
    int oc = nt * 16 + n;
    float bv_ = bias[oc];
#pragma unroll
    for (int r = 0; r < 8; ++r) {
      int mrow = row0 + wave * 16 + mh + r;
      outp[(size_t)mrow * DH + oc] = (__bf16)(acc[nt][r] + bv_);
    }
  }
}

// flog = clip(x.Wfi0+b,-5,5); i = exp(clip(x.Wfi1+b,-5,5)); one wave per row.
__global__ void gates_kernel(const float* __restrict__ x,
                             const float* __restrict__ wfi,
                             const float* __restrict__ bfi,
                             float* __restrict__ flog, float* __restrict__ iv) {
  const int D = 1024;
  int wave = threadIdx.x >> 5, l = threadIdx.x & 31;
  int row = blockIdx.x * 8 + wave;
  const float* xr = x + (size_t)row * D;
  float s0 = 0.f, s1 = 0.f;
  for (int j = l; j < D; j += 32) {
    float xv = xr[j];
    s0 += xv * wfi[j];
    s1 += xv * wfi[D + j];
  }
  for (int off = 16; off; off >>= 1) {
    s0 += __shfl_xor(s0, off, 32);
    s1 += __shfl_xor(s1, off, 32);
  }
  if (l == 0) {
    float d0 = fminf(fmaxf(s0 + bfi[0], -5.f), 5.f);
    float d1 = fminf(fmaxf(s1 + bfi[1], -5.f), 5.f);
    flog[row] = d0;
    iv[row] = expf(d1);
  }
}

// Per (batch,chunk): inclusive scans of log f and of w = i/F; kw = w*k (bf16).
__global__ void scan_kernel(const float* __restrict__ flog,
                            const float* __restrict__ iv,
                            const __bf16* __restrict__ kb,
                            float* __restrict__ cumlf, float* __restrict__ cumw,
                            __bf16* __restrict__ kwb, float* __restrict__ fprod,
                            float* __restrict__ wsum) {
  const int T = 128, DH = 128;
  __shared__ float s[128];
  int t = threadIdx.x;
  int idx = blockIdx.x;
  size_t pos = (size_t)idx * T + t;
  s[t] = flog[pos];
  __syncthreads();
  for (int off = 1; off < 128; off <<= 1) {
    float add = (t >= off) ? s[t - off] : 0.f;
    __syncthreads();
    s[t] += add;
    __syncthreads();
  }
  float clf = s[t];
  cumlf[pos] = clf;
  if (t == 127) fprod[idx] = expf(clf);
  float w = iv[pos] * expf(-clf);
  __syncthreads();
  s[t] = w;
  __syncthreads();
  for (int off = 1; off < 128; off <<= 1) {
    float add = (t >= off) ? s[t - off] : 0.f;
    __syncthreads();
    s[t] += add;
    __syncthreads();
  }
  cumw[pos] = s[t];
  if (t == 127) wsum[idx] = s[t];
  const __bf16* kr = kb + pos * DH;
  __bf16* kwr = kwb + pos * DH;
  for (int d = 0; d < DH; ++d) kwr[d] = (__bf16)(w * (float)kr[d]);
}

// G_c = V^T diag(w) K: both operands staged transposed through LDS so every
// fragment is two ds b128 loads.
__global__ void __launch_bounds__(256)
gmat_kernel(const __bf16* __restrict__ vb, const __bf16* __restrict__ kwb,
            float* __restrict__ G) {
  const int T = 128, DH = 128, SST = 136;
  __shared__ __align__(16) __bf16 VT[128 * 136];   // V^T  (d x s)
  __shared__ __align__(16) __bf16 KT_[128 * 136];  // Kw^T (n x s)
  int idx = blockIdx.x;
  int tid = threadIdx.x, wave = tid >> 5, l = tid & 31;
  size_t base = (size_t)idx * T;
  stage_transpose_128(vb + base * DH, VT, tid);
  stage_transpose_128(kwb + base * DH, KT_, tid);
  __syncthreads();
  int m0 = wave * 16;
  v8f acc[8];
#pragma unroll
  for (int i = 0; i < 8; ++i) acc[i] = zero8();
  for (int ks = 0; ks < T; ks += 32) {
    v16bf a = load_a_fast(VT + m0 * SST + ks, SST);
#pragma unroll
    for (int nt = 0; nt < 8; ++nt) {
      v16bf b = load_b_fast(KT_ + nt * 16 * SST + ks, SST);
      acc[nt] = wmma_bf16(a, b, acc[nt]);
    }
  }
  int mh = (l >> 4) * 8, n = l & 15;
  float* Gp = G + (size_t)idx * DH * DH;
#pragma unroll
  for (int nt = 0; nt < 8; ++nt)
#pragma unroll
    for (int r = 0; r < 8; ++r)
      Gp[(size_t)(m0 + mh + r) * DH + nt * 16 + n] = acc[nt][r];
}

// Sequential-over-chunks (NC=16) state chain: C_{c+1} = fprod*(C_c + G_c).
__global__ void chain_kernel(const float* __restrict__ G,
                             const float* __restrict__ fprod,
                             const float* __restrict__ wsum,
                             __bf16* __restrict__ Cbb, float* __restrict__ nb) {
  const int DH = 128, NC = 16;
  int b = blockIdx.x;
  int tid = threadIdx.x;
  float creg[64];
#pragma unroll
  for (int j = 0; j < 64; ++j) creg[j] = 0.f;
  float nrun = 0.f;
  for (int c = 0; c < NC; ++c) {
    int idx = b * NC + c;
    size_t off = (size_t)idx * DH * DH;
    float fp = fprod[idx];
#pragma unroll
    for (int j = 0; j < 64; ++j) {
      int e = tid + j * 256;
      Cbb[off + e] = (__bf16)creg[j];
      creg[j] = fp * (creg[j] + G[off + e]);
    }
    if (tid == 0) {
      nb[idx] = nrun;
      nrun = fp * (nrun + wsum[idx]);
    }
  }
}

// Per chunk: h = rowscale * (Q C_b^T + (mask o (Q Kw^T)) V)  -> bf16.
__global__ void __launch_bounds__(256)
intra_kernel(const __bf16* __restrict__ qb, const __bf16* __restrict__ kwb,
             const __bf16* __restrict__ vb, const __bf16* __restrict__ Cbb,
             const float* __restrict__ cumlf, const float* __restrict__ cumw,
             const float* __restrict__ nb, __bf16* __restrict__ hb) {
  const int T = 128, DH = 128, SST = 136;
  __shared__ __align__(16) __bf16 S[128 * 136];
  __shared__ __align__(16) __bf16 VT[128 * 136];  // V^T (n x s)
  __shared__ float rowscale[128];
  int idx = blockIdx.x;
  int tid = threadIdx.x, wave = tid >> 5, l = tid & 31;
  size_t base = (size_t)idx * T;
  unsigned int* su = (unsigned int*)S;
  for (int i = tid; i < 128 * 136 / 2; i += 256) su[i] = 0u;
  stage_transpose_128(vb + base * DH, VT, tid);
  if (tid < 128) {
    float clf = cumlf[base + tid];
    float ef = expf(clf);
    float nt = ef * (nb[idx] + cumw[base + tid]);
    rowscale[tid] = ef / fmaxf(fabsf(nt), 1.f);
  }
  __syncthreads();
  int m0 = wave * 16;
  // S = (Q Kw^T) with causal mask (kw already carries w_s = i_s/F_s).
  for (int j = 0; j <= wave; ++j) {
    v8f acc = zero8();
    for (int ks = 0; ks < DH; ks += 32) {
      v16bf a = load_a_fast(qb + (base + m0) * DH + ks, DH);
      v16bf b = load_b_fast(kwb + (base + j * 16) * DH + ks, DH);
      acc = wmma_bf16(a, b, acc);
    }
    int mh = (l >> 4) * 8, n = l & 15;
#pragma unroll
    for (int r = 0; r < 8; ++r) {
      int trow = m0 + mh + r, scol = j * 16 + n;
      float v = (scol <= trow) ? acc[r] : 0.f;
      S[trow * SST + scol] = (__bf16)v;
    }
  }
  __syncthreads();
  v8f acc2[8];
#pragma unroll
  for (int i = 0; i < 8; ++i) acc2[i] = zero8();
  // Inter-chunk: Q @ C_b^T (C_b rows contiguous -> fast B loads).
  const __bf16* Cp = Cbb + (size_t)idx * DH * DH;
  for (int ks = 0; ks < DH; ks += 32) {
    v16bf a = load_a_fast(qb + (base + m0) * DH + ks, DH);
#pragma unroll
    for (int nt = 0; nt < 8; ++nt) {
      v16bf b = load_b_fast(Cp + (size_t)(nt * 16) * DH + ks, DH);
      acc2[nt] = wmma_bf16(a, b, acc2[nt]);
    }
  }
  // Intra-chunk: S @ V, with V^T staged in LDS (fast B fragments).
  for (int ks = 0; ks < T; ks += 32) {
    v16bf a = load_a_fast(S + m0 * SST + ks, SST);
#pragma unroll
    for (int nt = 0; nt < 8; ++nt) {
      v16bf b = load_b_fast(VT + nt * 16 * SST + ks, SST);
      acc2[nt] = wmma_bf16(a, b, acc2[nt]);
    }
  }
  int mh = (l >> 4) * 8, n = l & 15;
#pragma unroll
  for (int nt = 0; nt < 8; ++nt)
#pragma unroll
    for (int r = 0; r < 8; ++r) {
      int trow = m0 + mh + r;
      float h = rowscale[trow] * acc2[nt][r];
      hb[(base + trow) * DH + nt * 16 + n] = (__bf16)h;
    }
}

// out = sigmoid(X Wo^T + bo) * (H Wup^T + bup).
// 128x64 output block per 256-thread block; X/Wo K-tiles staged into LDS by
// the Tensor Data Mover (double buffered, TENSORcnt), 16x64 strip per wave.
__global__ void __launch_bounds__(256)
final_kernel(const __bf16* __restrict__ xb, const __bf16* __restrict__ wob,
             const float* __restrict__ bo, const __bf16* __restrict__ hb,
             const __bf16* __restrict__ wupb, const float* __restrict__ bup,
             float* __restrict__ out) {
  const int D = 1024, DH = 128, KT = 64;
  __shared__ __align__(16) __bf16 Abuf[2][128 * KT];  // 2 x 16 KB
  __shared__ __align__(16) __bf16 Bbuf[2][64 * KT];   // 2 x 8 KB
  int tid = threadIdx.x, wave = tid >> 5, l = tid & 31;
  int bm = blockIdx.x >> 4, bn = blockIdx.x & 15;  // 128 x 16 blocks
  int row0 = bm * 128, col0 = bn * 64;

  stage_tile(xb + (size_t)row0 * D, Abuf[0], KT, 128, D, tid);
  stage_tile(wob + (size_t)col0 * D, Bbuf[0], KT, 64, D, tid);
  stage_commit(tid);

  v8f acc[4];
#pragma unroll
  for (int i = 0; i < 4; ++i) acc[i] = zero8();

  const int KSTEPS = D / KT;  // 16
  for (int ks = 0; ks < KSTEPS; ++ks) {
    int cur = ks & 1;
    if (ks + 1 < KSTEPS) {
      int kk = (ks + 1) * KT;
      stage_tile(xb + (size_t)row0 * D + kk, Abuf[cur ^ 1], KT, 128, D, tid);
      stage_tile(wob + (size_t)col0 * D + kk, Bbuf[cur ^ 1], KT, 64, D, tid);
    }
    const __bf16* Ab = Abuf[cur] + wave * 16 * KT;
#pragma unroll
    for (int k2 = 0; k2 < KT; k2 += 32) {
      v16bf a = load_a_fast(Ab + k2, KT);
#pragma unroll
      for (int nt = 0; nt < 4; ++nt) {
        v16bf b = load_b_fast(Bbuf[cur] + nt * 16 * KT + k2, KT);
        acc[nt] = wmma_bf16(a, b, acc[nt]);
      }
    }
    stage_commit(tid);
  }

  // Second GEMM: H(16384x128) @ Wup^T — small K, direct from global/L2.
  v8f acc2[4];
#pragma unroll
  for (int i = 0; i < 4; ++i) acc2[i] = zero8();
  for (int kk = 0; kk < DH; kk += 32) {
    v16bf a = load_a_fast(hb + (size_t)(row0 + wave * 16) * DH + kk, DH);
#pragma unroll
    for (int nt = 0; nt < 4; ++nt) {
      v16bf b = load_b_fast(wupb + (size_t)(col0 + nt * 16) * DH + kk, DH);
      acc2[nt] = wmma_bf16(a, b, acc2[nt]);
    }
  }

  int mh = (l >> 4) * 8, n = l & 15;
#pragma unroll
  for (int nt = 0; nt < 4; ++nt) {
    int col = col0 + nt * 16 + n;
    float bov = bo[col], bupv = bup[col];
#pragma unroll
    for (int r = 0; r < 8; ++r) {
      int row = row0 + wave * 16 + mh + r;
      float og = 1.f / (1.f + expf(-(acc[nt][r] + bov)));
      float u = acc2[nt][r] + bupv;
      out[(size_t)row * D + col] = og * u;
    }
  }
}

// ---------------------------------------------------------------------------
// Host launcher
// ---------------------------------------------------------------------------
extern "C" void kernel_launch(void* const* d_in, const int* in_sizes, int n_in,
                              void* d_out, int out_size, void* d_ws,
                              size_t ws_size, hipStream_t stream) {
  (void)in_sizes; (void)n_in; (void)out_size; (void)ws_size;
  const int Bb = 8, L = 2048, D = 1024, DH = 128, T = 128;
  const size_t N = (size_t)Bb * L;  // 16384 rows
  const int NC = L / T;             // 16 chunks per batch
  const int BC = Bb * NC;           // 128 (batch, chunk) pairs

  const float* x   = (const float*)d_in[0];
  const float* Wfi = (const float*)d_in[1];
  const float* bfi = (const float*)d_in[2];
  const float* Wo  = (const float*)d_in[3];
  const float* bo  = (const float*)d_in[4];
  const float* Wq  = (const float*)d_in[5];
  const float* bq  = (const float*)d_in[6];
  const float* Wk  = (const float*)d_in[7];
  const float* bk  = (const float*)d_in[8];
  const float* Wv  = (const float*)d_in[9];
  const float* bv  = (const float*)d_in[10];
  const float* Wup = (const float*)d_in[11];
  const float* bup = (const float*)d_in[12];

  char* ws = (char*)d_ws;
  size_t off = 0;
  auto take = [&](size_t bytes) -> char* {
    char* p = ws + off;
    off = (off + bytes + 255) & ~(size_t)255;
    return p;
  };
  __bf16* xb    = (__bf16*)take(N * D * 2);
  __bf16* wqkvb = (__bf16*)take((size_t)384 * D * 2);
  __bf16* wob   = (__bf16*)take((size_t)D * D * 2);
  __bf16* wupb  = (__bf16*)take((size_t)D * DH * 2);
  __bf16* qb    = (__bf16*)take(N * DH * 2);
  __bf16* kb    = (__bf16*)take(N * DH * 2);
  __bf16* vb    = (__bf16*)take(N * DH * 2);
  __bf16* kwb   = (__bf16*)take(N * DH * 2);
  float* flog   = (float*)take(N * 4);
  float* iv     = (float*)take(N * 4);
  float* cumlf  = (float*)take(N * 4);
  float* cumw   = (float*)take(N * 4);
  float* fprod  = (float*)take((size_t)BC * 4);
  float* wsum   = (float*)take((size_t)BC * 4);
  float* nbv    = (float*)take((size_t)BC * 4);
  float* G      = (float*)take((size_t)BC * DH * DH * 4);
  __bf16* Cbb   = (__bf16*)take((size_t)BC * DH * DH * 2);
  __bf16* hb    = (__bf16*)take(N * DH * 2);

  // fp32 -> bf16 conversions
  {
    int n = (int)(N * D);
    cvt_bf16_kernel<<<(n + 255) / 256, 256, 0, stream>>>(x, xb, n);
    int nw = 128 * D;
    cvt_bf16_kernel<<<(nw + 255) / 256, 256, 0, stream>>>(Wq, wqkvb, nw);
    cvt_bf16_kernel<<<(nw + 255) / 256, 256, 0, stream>>>(Wk, wqkvb + (size_t)128 * D, nw);
    cvt_bf16_kernel<<<(nw + 255) / 256, 256, 0, stream>>>(Wv, wqkvb + (size_t)256 * D, nw);
    int no = D * D;
    cvt_bf16_kernel<<<(no + 255) / 256, 256, 0, stream>>>(Wo, wob, no);
    int nu = D * DH;
    cvt_bf16_kernel<<<(nu + 255) / 256, 256, 0, stream>>>(Wup, wupb, nu);
  }

  // q/k/v projection: TDM-staged 128x128 blocks, 384 blocks total.
  proj_qkv_kernel<<<384, 256, 0, stream>>>(xb, wqkvb, bq, bk, bv, qb, kb, vb);
  // forget/input gates
  gates_kernel<<<(int)(N / 8), 256, 0, stream>>>(x, Wfi, bfi, flog, iv);
  // per-chunk scans + w-folded K
  scan_kernel<<<BC, 128, 0, stream>>>(flog, iv, kb, cumlf, cumw, kwb, fprod, wsum);
  // chunk outer-product state contributions (LDS-transposed operands)
  gmat_kernel<<<BC, 256, 0, stream>>>(vb, kwb, G);
  // inter-chunk state chain (16 sequential steps, parallel over state elems)
  chain_kernel<<<Bb, 256, 0, stream>>>(G, fprod, wsum, Cbb, nbv);
  // intra-chunk attention + inter-chunk matvec + normalization
  intra_kernel<<<BC, 256, 0, stream>>>(qb, kwb, vb, Cbb, cumlf, cumw, nbv, hb);
  // fused output GEMM, TDM-staged: 2048 blocks of 128x64 outputs.
  final_kernel<<<2048, 256, 0, stream>>>(xb, wob, bo, hb, wupb, bup,
                                         (float*)d_out);
}
